// MambaBlock_80805514707211
// MI455X (gfx1250) — matrix operand order repaired and checked by reading, not compiled, not run on hardware
//
#include <hip/hip_runtime.h>
#include <math.h>

// ---------------------------------------------------------------------------
// Mamba block forward for MI455X (gfx1250, wave32, WMMA).
// GEMMs use v_wmma_f32_16x16x32_bf16 (bf16 in, f32 accumulate).
// Dims: B=2 L=2048 H=1024 E=2048 N=16 R=64 K=4 ; M = B*L = 4096
// ---------------------------------------------------------------------------

#define Mrows 4096
#define Hc    1024
#define Ec    2048
#define Lc    2048
#define Nc    16
#define Rc    64
#define P2E   4096   // 2*E
#define PXN   96     // R + 2N

typedef __attribute__((ext_vector_type(16))) __bf16 v16bf;
typedef __attribute__((ext_vector_type(8)))  float  v8f;

__device__ __forceinline__ __bf16 f2bf(float f) { return (__bf16)f; }

union FragU { uint4 u[2]; v16bf v; };

// ---------------------------------------------------------------------------
// Large-N WMMA GEMM (N % 128 == 0): C[M,N] = A[M,K]*B[K,N], bf16 row-major.
// Block: 256 threads = 8 waves; block tile 64(M) x 128(N), K-step 32.
// Wave w: wm = w>>2 (0..1), wn = w&3 (0..3); each wave owns a 32x32 sub-tile
// (2 M-frags x 2 N-frags = 4 WMMAs / K-step, fragments reused).
// Double-buffered LDS; next tile's global loads overlap current compute.
// epilogue: 0 = +bias, 1 = softplus(x+bias), 2 = +bias+residual
// ---------------------------------------------------------------------------
__global__ void wmma_gemm_bf16_big(const __bf16* __restrict__ A,
                                   const __bf16* __restrict__ Bm,
                                   float* __restrict__ C,
                                   const float* __restrict__ bias,
                                   const float* __restrict__ resid,
                                   int M, int N, int K, int epilogue) {
  __shared__ __align__(16) __bf16 As[2][64 * 40];    // [m][k], stride 40
  __shared__ __align__(16) __bf16 Bs[2][128 * 40];   // [n][k] transposed

  const int tid  = threadIdx.x;
  const int m0   = blockIdx.y * 64;
  const int n0   = blockIdx.x * 128;

  const int lane = tid & 31;
  const int w    = tid >> 5;        // 0..7
  const int wm   = w >> 2;          // 0..1
  const int wn   = w & 3;           // 0..3
  const int half = lane >> 4;       // 0/1
  const int ml   = lane & 15;

  // --- staging addresses (advance by constants each K-step) ---
  const int aR = tid >> 2, aC = tid & 3;           // A: 64 rows x 4 groups
  const __bf16* aSrc = A + (size_t)(m0 + aR) * K + aC * 8;
  const int aDst = aR * 40 + aC * 8;
  const int bR = tid >> 4, bC = tid & 15;          // B: rows bR, bR+16 ; 16 grps
  const __bf16* bSrc0 = Bm + (size_t)bR * N + n0 + bC * 8;
  const __bf16* bSrc1 = bSrc0 + (size_t)16 * N;
  const size_t bStep = (size_t)32 * N;

  v8f acc[2][2] = {{{}, {}}, {{}, {}}};

  const int KT = K >> 5;

  // ---- prologue: stage tile 0 into buffer 0 ----
  {
    uint4 ra = *reinterpret_cast<const uint4*>(aSrc);
    union { uint4 u; __bf16 h[8]; } b0, b1;
    b0.u = *reinterpret_cast<const uint4*>(bSrc0);
    b1.u = *reinterpret_cast<const uint4*>(bSrc1);
    aSrc += 32; bSrc0 += bStep; bSrc1 += bStep;
    *reinterpret_cast<uint4*>(&As[0][aDst]) = ra;
#pragma unroll
    for (int j = 0; j < 8; ++j) {
      Bs[0][(bC * 8 + j) * 40 + bR]      = b0.h[j];
      Bs[0][(bC * 8 + j) * 40 + 16 + bR] = b1.h[j];
    }
  }
  __syncthreads();

  for (int kt = 0; kt < KT; ++kt) {
    const int cur = kt & 1, nxt = cur ^ 1;
    const bool hasNext = (kt + 1 < KT);

    // ---- issue next tile's global loads (latency overlapped) ----
    uint4 ra;
    union { uint4 u; __bf16 h[8]; } b0, b1;
    if (hasNext) {
      ra   = *reinterpret_cast<const uint4*>(aSrc);
      b0.u = *reinterpret_cast<const uint4*>(bSrc0);
      b1.u = *reinterpret_cast<const uint4*>(bSrc1);
      aSrc += 32; bSrc0 += bStep; bSrc1 += bStep;
      __builtin_prefetch(aSrc, 0, 0);              // global_prefetch_b8 (k+2)
      __builtin_prefetch(bSrc0, 0, 0);
    }

    // ---- compute current buffer: 2 A frags, 2 B frags, 4 WMMAs ----
    FragU fa[2], fb[2];
#pragma unroll
    for (int im = 0; im < 2; ++im) {
      int aoff = (wm * 32 + im * 16 + ml) * 40 + half * 8;
      fa[im].u[0] = *reinterpret_cast<const uint4*>(&As[cur][aoff]);
      fa[im].u[1] = *reinterpret_cast<const uint4*>(&As[cur][aoff + 16]);
    }
#pragma unroll
    for (int in_ = 0; in_ < 2; ++in_) {
      int boff = (wn * 32 + in_ * 16 + ml) * 40 + half * 8;
      fb[in_].u[0] = *reinterpret_cast<const uint4*>(&Bs[cur][boff]);
      fb[in_].u[1] = *reinterpret_cast<const uint4*>(&Bs[cur][boff + 16]);
    }
#pragma unroll
    for (int im = 0; im < 2; ++im)
#pragma unroll
      for (int in_ = 0; in_ < 2; ++in_)
        acc[im][in_] = __builtin_amdgcn_wmma_f32_16x16x32_bf16(
            false, fa[im].v, false, fb[in_].v,
            (short)0, acc[im][in_], false, false);

    __syncthreads();   // prior readers of buf[nxt] are done
    if (hasNext) {
      *reinterpret_cast<uint4*>(&As[nxt][aDst]) = ra;
#pragma unroll
      for (int j = 0; j < 8; ++j) {
        Bs[nxt][(bC * 8 + j) * 40 + bR]      = b0.h[j];
        Bs[nxt][(bC * 8 + j) * 40 + 16 + bR] = b1.h[j];
      }
      __syncthreads(); // stores visible before next iteration's reads
    }
  }

  // ---- epilogue ----
#pragma unroll
  for (int in_ = 0; in_ < 2; ++in_) {
    const int col  = n0 + wn * 32 + in_ * 16 + ml;
    const float bcol = bias ? bias[col] : 0.0f;
#pragma unroll
    for (int im = 0; im < 2; ++im) {
#pragma unroll
      for (int i = 0; i < 8; ++i) {
        int row = m0 + wm * 32 + im * 16 + half * 8 + i;
        float v = acc[im][in_][i] + bcol;
        if (epilogue == 1) {
          v = (v > 20.0f) ? v : log1pf(__expf(v));
        } else if (epilogue == 2) {
          v += resid[(size_t)row * N + col];
        }
        C[(size_t)row * N + col] = v;
      }
    }
  }
}

// ---------------------------------------------------------------------------
// Small-N WMMA GEMM (N % 32 == 0) — used for x_proj (N=96).
// Block 256 threads = 8 waves; tile 64x32; each wave one 16x16 fragment.
// ---------------------------------------------------------------------------
__global__ void wmma_gemm_bf16_small(const __bf16* __restrict__ A,
                                     const __bf16* __restrict__ Bm,
                                     float* __restrict__ C,
                                     const float* __restrict__ bias,
                                     int M, int N, int K) {
  __shared__ __align__(16) __bf16 As[64 * 40];
  __shared__ __align__(16) __bf16 Bs[32 * 40];

  const int tid  = threadIdx.x;
  const int m0   = blockIdx.y * 64;
  const int n0   = blockIdx.x * 32;
  const int lane = tid & 31;
  const int w    = tid >> 5;
  const int wm   = w >> 1;          // 0..3
  const int wn   = w & 1;           // 0..1
  const int half = lane >> 4;
  const int ml   = lane & 15;

  const int aR = tid >> 2, aC = tid & 3;
  const __bf16* aSrc = A + (size_t)(m0 + aR) * K + aC * 8;
  const int aDst = aR * 40 + aC * 8;
  const int bR = tid >> 2, bC = tid & 3;           // threads < 128
  const __bf16* bSrc = Bm + (size_t)bR * N + n0 + bC * 8;
  const size_t bStep = (size_t)32 * N;

  v8f acc = {};
  for (int k0 = 0; k0 < K; k0 += 32) {
    *reinterpret_cast<uint4*>(&As[aDst]) = *reinterpret_cast<const uint4*>(aSrc);
    aSrc += 32;
    if (tid < 128) {
      union { uint4 u; __bf16 h[8]; } bv;
      bv.u = *reinterpret_cast<const uint4*>(bSrc);
#pragma unroll
      for (int j = 0; j < 8; ++j) Bs[(bC * 8 + j) * 40 + bR] = bv.h[j];
    }
    bSrc += bStep;
    __syncthreads();

    FragU fa, fb;
    int aoff = (wm * 16 + ml) * 40 + half * 8;
    fa.u[0] = *reinterpret_cast<const uint4*>(&As[aoff]);
    fa.u[1] = *reinterpret_cast<const uint4*>(&As[aoff + 16]);
    int boff = (wn * 16 + ml) * 40 + half * 8;
    fb.u[0] = *reinterpret_cast<const uint4*>(&Bs[boff]);
    fb.u[1] = *reinterpret_cast<const uint4*>(&Bs[boff + 16]);

    acc = __builtin_amdgcn_wmma_f32_16x16x32_bf16(
        false, fa.v, false, fb.v, (short)0, acc, false, false);
    __syncthreads();
  }

  const int col = n0 + wn * 16 + ml;
  const float bcol = bias ? bias[col] : 0.0f;
#pragma unroll
  for (int i = 0; i < 8; ++i) {
    int row = m0 + wm * 16 + half * 8 + i;
    C[(size_t)row * N + col] = acc[i] + bcol;
  }
}

// ---------------------------------------------------------------------------
// LayerNorm over H, writes bf16 for the in_proj GEMM. One block per row.
// ---------------------------------------------------------------------------
__global__ void ln_kernel(const float* __restrict__ x,
                          const float* __restrict__ w,
                          const float* __restrict__ b,
                          __bf16* __restrict__ out) {
  const int m = blockIdx.x;
  const float* row = x + (size_t)m * Hc;
  float s = 0.f, s2 = 0.f;
  for (int h = threadIdx.x; h < Hc; h += blockDim.x) {
    float v = row[h]; s += v; s2 += v * v;
  }
  __shared__ float rs[256], rq[256];
  rs[threadIdx.x] = s; rq[threadIdx.x] = s2;
  __syncthreads();
  for (int o = 128; o > 0; o >>= 1) {
    if (threadIdx.x < o) { rs[threadIdx.x] += rs[threadIdx.x + o];
                           rq[threadIdx.x] += rq[threadIdx.x + o]; }
    __syncthreads();
  }
  const float mu  = rs[0] * (1.0f / Hc);
  const float var = rq[0] * (1.0f / Hc) - mu * mu;
  const float inv = rsqrtf(var + 1e-5f);
  for (int h = threadIdx.x; h < Hc; h += blockDim.x)
    out[(size_t)m * Hc + h] = f2bf((row[h] - mu) * inv * w[h] + b[h]);
}

// ---------------------------------------------------------------------------
// Causal depthwise conv (K=4) + SiLU.  x comes from proj[:, 0:E] (stride 2E).
// ---------------------------------------------------------------------------
__global__ void conv_silu_kernel(const float* __restrict__ proj,
                                 const float* __restrict__ cw,
                                 const float* __restrict__ cb,
                                 float* __restrict__ xf,
                                 __bf16* __restrict__ xb) {
  const int i = blockIdx.x * blockDim.x + threadIdx.x;   // over M*E
  const int e = i & (Ec - 1);
  const int m = i / Ec;
  const int bb = m >> 11;
  const int l  = m & (Lc - 1);
  float acc = cb[e];
#pragma unroll
  for (int k = 0; k < 4; ++k) {
    int li = l - 3 + k;
    if (li >= 0)
      acc += proj[((size_t)(bb * Lc + li)) * P2E + e] * cw[e * 4 + k];
  }
  float sv = acc / (1.0f + __expf(-acc));
  xf[i] = sv;
  xb[i] = f2bf(sv);
}

// ---------------------------------------------------------------------------
// Selective scan. One thread per (b,e); h[16] in registers; B_t/C_t vectors
// double-buffered in LDS (one barrier per timestep).
// ---------------------------------------------------------------------------
__global__ void scan_kernel(const float* __restrict__ xconv,
                            const float* __restrict__ dt,
                            const float* __restrict__ dbc,
                            const float* __restrict__ A_log,
                            const float* __restrict__ Dp,
                            float* __restrict__ y) {
  const int bb = blockIdx.x >> 3;                      // E/256 = 8 chunks
  const int e  = ((blockIdx.x & 7) << 8) + threadIdx.x;
  float a[Nc];
#pragma unroll
  for (int n = 0; n < Nc; ++n) a[n] = -__expf(A_log[e * Nc + n]);
  const float Dv = Dp[e];
  float h[Nc];
#pragma unroll
  for (int n = 0; n < Nc; ++n) h[n] = 0.0f;

  __shared__ float sBC[2][32];                         // [0..15]=B_t [16..31]=C_t
  const size_t rowBase = (size_t)bb * Lc;
  if (threadIdx.x < 32) sBC[0][threadIdx.x] = dbc[rowBase * PXN + Rc + threadIdx.x];
  __syncthreads();
  for (int t = 0; t < Lc; ++t) {
    const size_t row = rowBase + t;
    const int cur = t & 1;
    if (threadIdx.x < 32 && t + 1 < Lc)
      sBC[cur ^ 1][threadIdx.x] = dbc[(row + 1) * PXN + Rc + threadIdx.x];
    const float dtv = dt[row * Ec + e];
    const float xv  = xconv[row * Ec + e];
    float yv = 0.0f;
#pragma unroll
    for (int n = 0; n < Nc; ++n) {
      float dA = __expf(dtv * a[n]);
      h[n] = dA * h[n] + dtv * sBC[cur][n] * xv;
      yv += h[n] * sBC[cur][16 + n];
    }
    y[row * Ec + e] = yv + xv * Dv;
    __syncthreads();
  }
}

// ---------------------------------------------------------------------------
// Gate: y = sigmoid(gate) * y, emit bf16 for the out_proj GEMM.
// ---------------------------------------------------------------------------
__global__ void gate_kernel(const float* __restrict__ proj,
                            const float* __restrict__ y,
                            __bf16* __restrict__ yb) {
  const int i = blockIdx.x * blockDim.x + threadIdx.x;   // over M*E
  const int e = i & (Ec - 1);
  const int m = i / Ec;
  const float g  = proj[(size_t)m * P2E + Ec + e];
  const float yv = y[i];
  yb[i] = f2bf(yv / (1.0f + __expf(-g)));
}

// ---------------------------------------------------------------------------
// Converters.
// ---------------------------------------------------------------------------
__global__ void f32_to_bf16_kernel(const float* __restrict__ s,
                                   __bf16* __restrict__ d, int n) {
  int i = blockIdx.x * blockDim.x + threadIdx.x;
  if (i < n) d[i] = f2bf(s[i]);
}

__global__ void dtr_convert_kernel(const float* __restrict__ dbc,
                                   __bf16* __restrict__ d) {
  int i = blockIdx.x * blockDim.x + threadIdx.x;       // over M*64
  int m = i >> 6, c = i & 63;
  d[i] = f2bf(dbc[(size_t)m * PXN + c]);
}

// ---------------------------------------------------------------------------
extern "C" void kernel_launch(void* const* d_in, const int* in_sizes, int n_in,
                              void* d_out, int out_size, void* d_ws, size_t ws_size,
                              hipStream_t stream) {
  const float* hidden   = (const float*)d_in[0];
  const float* ln_w     = (const float*)d_in[1];
  const float* ln_b     = (const float*)d_in[2];
  const float* inproj_w = (const float*)d_in[3];
  const float* inproj_b = (const float*)d_in[4];
  const float* conv_w   = (const float*)d_in[5];
  const float* conv_b   = (const float*)d_in[6];
  const float* xproj_w  = (const float*)d_in[7];
  const float* dtproj_w = (const float*)d_in[8];
  const float* dtproj_b = (const float*)d_in[9];
  const float* A_log    = (const float*)d_in[10];
  const float* Dp       = (const float*)d_in[11];
  const float* outproj_w= (const float*)d_in[12];
  const float* outproj_b= (const float*)d_in[13];
  float* out            = (float*)d_out;

  char* p = (char*)d_ws;
  auto alloc = [&](size_t bytes) {
    char* r = p; p += (bytes + 255) & ~(size_t)255; return (void*)r;
  };
  __bf16* hs_bf    = (__bf16*)alloc((size_t)Mrows * Hc * 2);
  __bf16* win_bf   = (__bf16*)alloc((size_t)Hc * P2E * 2);
  float*  proj     = (float*) alloc((size_t)Mrows * P2E * 4);
  float*  xconv_f  = (float*) alloc((size_t)Mrows * Ec * 4);
  __bf16* xconv_bf = (__bf16*)alloc((size_t)Mrows * Ec * 2);
  __bf16* wxp_bf   = (__bf16*)alloc((size_t)Ec * PXN * 2);
  float*  dbc      = (float*) alloc((size_t)Mrows * PXN * 4);
  __bf16* dtr_bf   = (__bf16*)alloc((size_t)Mrows * Rc * 2);
  __bf16* wdt_bf   = (__bf16*)alloc((size_t)Rc * Ec * 2);
  float*  dtbuf    = (float*) alloc((size_t)Mrows * Ec * 4);
  float*  yscan    = (float*) alloc((size_t)Mrows * Ec * 4);
  __bf16* y_bf     = (__bf16*)alloc((size_t)Mrows * Ec * 2);
  __bf16* wout_bf  = (__bf16*)alloc((size_t)Ec * Hc * 2);

  // 1. LayerNorm -> bf16
  ln_kernel<<<Mrows, 256, 0, stream>>>(hidden, ln_w, ln_b, hs_bf);

  // 2. weight conversions f32 -> bf16
  {
    int n;
    n = Hc * P2E; f32_to_bf16_kernel<<<(n + 255) / 256, 256, 0, stream>>>(inproj_w, win_bf, n);
    n = Ec * PXN; f32_to_bf16_kernel<<<(n + 255) / 256, 256, 0, stream>>>(xproj_w,  wxp_bf, n);
    n = Rc * Ec;  f32_to_bf16_kernel<<<(n + 255) / 256, 256, 0, stream>>>(dtproj_w, wdt_bf, n);
    n = Ec * Hc;  f32_to_bf16_kernel<<<(n + 255) / 256, 256, 0, stream>>>(outproj_w, wout_bf, n);
  }

  // 3. in_proj GEMM: (M x H) @ (H x 2E) + bias
  wmma_gemm_bf16_big<<<dim3(P2E / 128, Mrows / 64), 256, 0, stream>>>(
      hs_bf, win_bf, proj, inproj_b, nullptr, Mrows, P2E, Hc, 0);

  // 4. causal depthwise conv + SiLU
  conv_silu_kernel<<<(size_t)Mrows * Ec / 256, 256, 0, stream>>>(
      proj, conv_w, conv_b, xconv_f, xconv_bf);

  // 5. x_proj GEMM: (M x E) @ (E x 96), no bias
  wmma_gemm_bf16_small<<<dim3(PXN / 32, Mrows / 64), 256, 0, stream>>>(
      xconv_bf, wxp_bf, dbc, nullptr, Mrows, PXN, Ec);

  // 6. dt_r slice -> bf16
  dtr_convert_kernel<<<(size_t)Mrows * Rc / 256, 256, 0, stream>>>(dbc, dtr_bf);

  // 7. dt_proj GEMM: (M x 64) @ (64 x E) + bias, softplus epilogue
  wmma_gemm_bf16_big<<<dim3(Ec / 128, Mrows / 64), 256, 0, stream>>>(
      dtr_bf, wdt_bf, dtbuf, dtproj_b, nullptr, Mrows, Ec, Rc, 1);

  // 8. selective scan (sequential over L, parallel over B*E)
  scan_kernel<<<16, 256, 0, stream>>>(xconv_f, dtbuf, dbc, A_log, Dp, yscan);

  // 9. gate: sigmoid(gate) * y -> bf16
  gate_kernel<<<(size_t)Mrows * Ec / 256, 256, 0, stream>>>(proj, yscan, y_bf);

  // 10. out_proj GEMM: (M x E) @ (E x H) + bias + residual -> d_out (f32)
  wmma_gemm_bf16_big<<<dim3(Hc / 128, Mrows / 64), 256, 0, stream>>>(
      y_bf, wout_bf, out, outproj_b, hidden, Mrows, Hc, Ec, 2);

  (void)in_sizes; (void)n_in; (void)out_size; (void)ws_size;
}
